// Model_28930899705876
// MI455X (gfx1250) — compile-verified, matrix operand order checked by reading, not run on hardware
//
#include <hip/hip_runtime.h>
#include <hip/hip_bf16.h>
#include <math.h>

// ---------------- problem constants (match reference) ----------------
#define NN     40000
#define DEG    16
#define EE     (NN * DEG)
#define D0     64
#define DV     16
#define DIN    112          // D0 + 3*DV
#define DM     128
#define NRBF   16
#define H_R    32
#define MAX_RIGID 9
#define MAX_ATOM  24

typedef __bf16 bf16;
typedef __attribute__((ext_vector_type(16))) __bf16 bf16x16;
typedef __attribute__((ext_vector_type(8)))  __bf16 bf16x8;
typedef __attribute__((ext_vector_type(8)))  float  f32x8;

// =====================================================================
// Kernel 1: per-edge RBF features  rbf[e][k] = exp(-2*(d - c_k)^2)
// =====================================================================
__global__ void k_rbf(const float* __restrict__ pos,
                      const int* __restrict__ esrc, const int* __restrict__ edst,
                      float* __restrict__ rbf, int ne) {
    int e = blockIdx.x * blockDim.x + threadIdx.x;
    if (e >= ne) return;
    int s = esrc[e], d = edst[e];
    float dx = pos[d*3+0] - pos[s*3+0];
    float dy = pos[d*3+1] - pos[s*3+1];
    float dz = pos[d*3+2] - pos[s*3+2];
    float dist = sqrtf(dx*dx + dy*dy + dz*dz + 1e-12f);
    float* out = rbf + (size_t)e * NRBF;
#pragma unroll
    for (int k = 0; k < NRBF; ++k) {
        float c = (8.0f / 15.0f) * (float)k;
        float t = dist - c;
        out[k] = __expf(-2.0f * t * t);
    }
}

// =====================================================================
// Kernel 2: build bf16 input features, zero-padded to 128 columns
// =====================================================================
__global__ void k_build_fin(const float* __restrict__ f0,
                            const float* __restrict__ f1,
                            bf16* __restrict__ out, int n) {
    int i = blockIdx.x * blockDim.x + threadIdx.x;
    if (i >= n * 128) return;
    int r = i >> 7, c = i & 127;
    float v = 0.0f;
    if (c < D0)        v = f0[(size_t)r * D0 + c];
    else if (c < DIN)  v = f1[(size_t)r * (3*DV) + (c - D0)];
    out[i] = (bf16)v;
}

// =====================================================================
// Kernel 3: pack fp32 weight (din x dout, row-major) into the exact
// per-lane bf16 B-fragment order of v_wmma_f32_16x16x32_bf16.
// Layout: wp[((kt*ntiles + tn)*32 + lane)*16 + j]
//   lane: g=lane>>4, n=lane&15 -> column tn*16+n
//   j<8 : K = kt*32 + g*8 + j ;  j>=8 : K = kt*32 + 16 + g*8 + (j-8)
// Zero-pads K>=din and col>=dout.
// =====================================================================
__global__ void k_pack_w(const float* __restrict__ W, bf16* __restrict__ wp,
                         int din, int dout, int ktiles, int ntiles) {
    int t = blockIdx.x * blockDim.x + threadIdx.x;
    int total = ktiles * ntiles * 32;
    if (t >= total) return;
    int lane = t & 31;
    int tile = t >> 5;
    int tn = tile % ntiles;
    int kt = tile / ntiles;
    int g = lane >> 4;
    int col = tn * 16 + (lane & 15);
    bf16x16 out;
#pragma unroll
    for (int j = 0; j < 16; ++j) {
        int k = kt * 32 + ((j < 8) ? (g * 8 + j) : (16 + g * 8 + (j - 8)));
        float v = (k < din && col < dout) ? W[(size_t)k * dout + col] : 0.0f;
        out[j] = (bf16)v;
    }
    *((bf16x16*)wp + ((size_t)tile * 32 + lane)) = out;
}

// =====================================================================
// Kernel 4: dual WMMA GEMM.  One wave computes a 16x16 tile of BOTH
//   M   = relu(feat @ W_msg)            (bf16, stride dout_t)
//   agg = feat @ W_self + b             (fp32, stride dout_t)
// A-fragments are shared between the two accumulators.
// feat is N x 128 bf16 (zero-padded); ktiles = 4.
// =====================================================================
__global__ void __launch_bounds__(256)
k_gemm_dual(const bf16* __restrict__ feat,
            const bf16* __restrict__ wpA, const bf16* __restrict__ wpB,
            const float* __restrict__ bias, int dout,
            bf16* __restrict__ Mout, float* __restrict__ agg,
            int nrows, int dout_t, int ktiles) {
    int wave = (int)((blockIdx.x * blockDim.x + threadIdx.x) >> 5);
    int lane = threadIdx.x & 31;
    int ntiles = dout_t >> 4;
    int total = (nrows >> 4) * ntiles;
    if (wave >= total) return;
    int tm = wave / ntiles;
    int tn = wave % ntiles;
    int g = lane >> 4;
    int m = lane & 15;

    f32x8 accA = {};
    f32x8 accB = {};
    const bf16* arow = feat + (size_t)(tm * 16 + m) * 128;

    for (int kt = 0; kt < ktiles; ++kt) {
        // A fragment: two contiguous 16B chunks of this lane's row
        const bf16* ab = arow + kt * 32 + g * 8;
        bf16x8 lo = *(const bf16x8*)(ab);
        bf16x8 hi = *(const bf16x8*)(ab + 16);
        bf16x16 a;
#pragma unroll
        for (int i = 0; i < 8; ++i) { a[i] = lo[i]; a[8 + i] = hi[i]; }
        // B fragments: pre-packed, fully contiguous per lane
        size_t boff = ((size_t)(kt * ntiles + tn) * 32 + lane) * 16;
        bf16x16 bA = *(const bf16x16*)(wpA + boff);
        bf16x16 bB = *(const bf16x16*)(wpB + boff);
#if defined(__HIP_DEVICE_COMPILE__)
        accA = __builtin_amdgcn_wmma_f32_16x16x32_bf16(
            false, a, false, bA, (short)0, accA, false, false);
        accB = __builtin_amdgcn_wmma_f32_16x16x32_bf16(
            false, a, false, bB, (short)0, accB, false, false);
#endif
    }

    int col = tn * 16 + m;
    float bv = (col < dout) ? bias[col] : 0.0f;
#pragma unroll
    for (int r = 0; r < 8; ++r) {
        int row = tm * 16 + r + 8 * g;           // C layout: VGPR r, lane group g
        float sA = accA[r];
        Mout[(size_t)row * dout_t + col] = (bf16)(sA > 0.0f ? sA : 0.0f);
        agg [(size_t)row * dout_t + col] = accB[r] + bv;
    }
}

// =====================================================================
// Kernel 5: per-edge radial MLP + gather(src) * radial + scatter-add(dst)
// One wave per edge; radial weights cached in LDS; h broadcast via shfl.
// radial = relu(rbf @ w_r1 + b_r1) @ w_r2 + b_r2
// agg[dst] += (float)M[src] * radial     (global_atomic_add_f32)
// =====================================================================
__global__ void __launch_bounds__(256)
k_edge(const float* __restrict__ rbf,
       const int* __restrict__ esrc, const int* __restrict__ edst,
       const bf16* __restrict__ Mbf, float* __restrict__ agg,
       const float* __restrict__ wr1, const float* __restrict__ br1,
       const float* __restrict__ wr2, const float* __restrict__ br2,
       int nedges, int dout, int dout_t) {
    __shared__ float sW1[NRBF * H_R];   // 512
    __shared__ float sB1[H_R];
    __shared__ float sW2[H_R * 128];    // up to 4096
    __shared__ float sB2[128];
    for (int i = threadIdx.x; i < NRBF * H_R; i += blockDim.x) sW1[i] = wr1[i];
    for (int i = threadIdx.x; i < H_R;        i += blockDim.x) sB1[i] = br1[i];
    for (int i = threadIdx.x; i < H_R * dout; i += blockDim.x) sW2[i] = wr2[i];
    for (int i = threadIdx.x; i < dout;       i += blockDim.x) sB2[i] = br2[i];
    __syncthreads();

    int lane   = threadIdx.x & 31;
    int wave   = (int)((blockIdx.x * blockDim.x + threadIdx.x) >> 5);
    int nwaves = (int)((gridDim.x * blockDim.x) >> 5);
    int nch    = (dout + 31) >> 5;

    for (int e = wave; e < nedges; e += nwaves) {
        int s = esrc[e], d = edst[e];
        // prefetch next iteration's rbf row (speculative, gfx1250 global_prefetch)
        __builtin_prefetch(rbf + (size_t)(e + nwaves) * NRBF, 0, 0);

        float rv = (lane < NRBF) ? rbf[(size_t)e * NRBF + lane] : 0.0f;
        // hidden: lane l computes h[l]
        float acc = sB1[lane];
#pragma unroll
        for (int k = 0; k < NRBF; ++k) {
            float rk = __shfl(rv, k, 32);
            acc += rk * sW1[k * H_R + lane];
        }
        float h = acc > 0.0f ? acc : 0.0f;

        float rad[4] = {0.f, 0.f, 0.f, 0.f};
#pragma unroll
        for (int j = 0; j < H_R; ++j) {
            float hj = __shfl(h, j, 32);
            for (int ci = 0; ci < nch; ++ci) {
                int c = lane + (ci << 5);
                rad[ci] += hj * sW2[j * dout + c];   // in-bounds of sW2 array
            }
        }
        const bf16* mrow = Mbf + (size_t)s * dout_t;
        float*      arow = agg + (size_t)d * dout_t;
        for (int ci = 0; ci < nch; ++ci) {
            int c = lane + (ci << 5);
            if (c < dout) {
                float mv = (float)mrow[c] * (rad[ci] + sB2[c]);
                unsafeAtomicAdd(&arow[c], mv);
            }
        }
    }
}

// =====================================================================
// Kernel 6: post  out = (act ? relu(agg) : agg) + (res ? res : 0)
// writes fp32 always, bf16 copy optionally (next layer's A operand)
// =====================================================================
__global__ void k_post(const float* __restrict__ agg, const float* __restrict__ res,
                       float* __restrict__ outf, bf16* __restrict__ outbf,
                       int n, int dout_t, int act) {
    int i = blockIdx.x * blockDim.x + threadIdx.x;
    if (i >= n * dout_t) return;
    float v = agg[i];
    if (act) v = v > 0.0f ? v : 0.0f;
    if (res) v += res[i];
    outf[i] = v;
    if (outbf) outbf[i] = (bf16)v;
}

// =====================================================================
// Kernel 7: rigid-body kinematics tail (one thread per node)
// =====================================================================
__device__ __forceinline__ void combine4x3(const float X[4][3], const float Y[4][3],
                                           float O[4][3]) {
    float T[4][3];
#pragma unroll
    for (int i = 0; i < 3; ++i)
#pragma unroll
        for (int k = 0; k < 3; ++k)
            T[i][k] = X[i][0]*Y[0][k] + X[i][1]*Y[1][k] + X[i][2]*Y[2][k];
#pragma unroll
    for (int i = 0; i < 3; ++i)
        T[3][i] = X[i][0]*Y[3][0] + X[i][1]*Y[3][1] + X[i][2]*Y[3][2] + X[3][i];
#pragma unroll
    for (int i = 0; i < 4; ++i)
#pragma unroll
        for (int k = 0; k < 3; ++k)
            O[i][k] = T[i][k];
}

__global__ void k_rigid(const float* __restrict__ bbv, const float* __restrict__ scv,
                        const float* __restrict__ pos, const int* __restrict__ rtype,
                        const float* __restrict__ rgt, const int* __restrict__ rgd,
                        const float* __restrict__ rtt, const int* __restrict__ rtd,
                        float* __restrict__ out, int n) {
    int i = blockIdx.x * blockDim.x + threadIdx.x;
    if (i >= n) return;
    int rt = rtype[i];
    const float* bb = bbv + (size_t)i * 16;
    const float* sc = scv + (size_t)i * 16;

    float qw = 1.0f, qx = bb[0], qy = bb[1], qz = bb[2];
    float qn = rsqrtf(qw*qw + qx*qx + qy*qy + qz*qz);
    qw *= qn; qx *= qn; qy *= qn; qz *= qn;

    float opr[MAX_RIGID][4][3];
    // rigid 0: quaternion rotation + translation bb[3:6] + pos
    opr[0][0][0] = qw*qw + qx*qx - qy*qy - qz*qz;
    opr[0][0][1] = 2.f*(qx*qy - qw*qz);
    opr[0][0][2] = 2.f*(qx*qz + qw*qy);
    opr[0][1][0] = 2.f*(qx*qy + qw*qz);
    opr[0][1][1] = qw*qw - qx*qx + qy*qy - qz*qz;
    opr[0][1][2] = 2.f*(qy*qz - qw*qx);
    opr[0][2][0] = 2.f*(qx*qz - qw*qy);
    opr[0][2][1] = 2.f*(qy*qz + qw*qx);
    opr[0][2][2] = qw*qw - qx*qx - qy*qy + qz*qz;
    opr[0][3][0] = bb[3] + pos[i*3+0];
    opr[0][3][1] = bb[4] + pos[i*3+1];
    opr[0][3][2] = bb[5] + pos[i*3+2];

    // rigids 1..8: X-axis torsion rotations
    for (int r = 1; r < MAX_RIGID; ++r) {
        float s0 = sc[(r-1)*2 + 0], c0 = sc[(r-1)*2 + 1];
        float inv = rsqrtf(s0*s0 + c0*c0);
        float sn = s0 * inv, cs = c0 * inv;
#pragma unroll
        for (int a = 0; a < 4; ++a)
#pragma unroll
            for (int b = 0; b < 3; ++b) opr[r][a][b] = 0.0f;
        opr[r][0][0] = 1.0f;
        opr[r][1][1] = cs;  opr[r][1][2] = -sn;
        opr[r][2][1] = sn;  opr[r][2][2] = cs;
    }

    // opr = combine(transforms[rt], opr)
    const float* T = rtt + (size_t)rt * MAX_RIGID * 12;
    for (int r = 0; r < MAX_RIGID; ++r) {
        float X[4][3];
#pragma unroll
        for (int a = 0; a < 4; ++a)
#pragma unroll
            for (int b = 0; b < 3; ++b) X[a][b] = T[r*12 + a*3 + b];
        combine4x3(X, opr[r], opr[r]);
    }

    // dependency chain (deps strictly < r, already finalized)
    const int* tdep = rtd + rt * MAX_RIGID;
    for (int r = 1; r < MAX_RIGID; ++r) {
        int dp = tdep[r];
        combine4x3(opr[dp], opr[r], opr[r]);
    }

    // atoms
    const float* G  = rgt + (size_t)rt * MAX_ATOM * 3;
    const int*   gd = rgd + rt * MAX_ATOM;
    float* o = out + (size_t)i * MAX_ATOM * 3;
    for (int a = 0; a < MAX_ATOM; ++a) {
        int g = gd[a];
        float vx = G[a*3+0], vy = G[a*3+1], vz = G[a*3+2];
#pragma unroll
        for (int c = 0; c < 3; ++c)
            o[a*3+c] = opr[g][c][0]*vx + opr[g][c][1]*vy + opr[g][c][2]*vz + opr[g][3][c];
    }
}

// =====================================================================
// Host orchestration
// =====================================================================
struct Lay {
    const float *Wm, *Ws, *b, *br1, *br2, *wr1, *wr2;
    int din, dout;
};

extern "C" void kernel_launch(void* const* d_in, const int* in_sizes, int n_in,
                              void* d_out, int out_size, void* d_ws, size_t ws_size,
                              hipStream_t stream) {
    (void)in_sizes; (void)n_in; (void)out_size; (void)ws_size;
    const float* f_in0 = (const float*)d_in[0];
    const float* f_in1 = (const float*)d_in[1];
    const float* pos   = (const float*)d_in[2];
    const int*   esrc  = (const int*)  d_in[3];
    const int*   edst  = (const int*)  d_in[4];
    const int*   rtype = (const int*)  d_in[5];
    const float* P     = (const float*)d_in[6];
    const float* rgt   = (const float*)d_in[7];
    const int*   rgd   = (const int*)  d_in[8];
    const float* rtt   = (const float*)d_in[9];
    const int*   rtd   = (const int*)  d_in[10];

    // ---- workspace carving (byte offsets, all 256B aligned) ----
    char* ws = (char*)d_ws;
    float* rbf     = (float*)(ws + 0);                        // E*16*4   = 40,960,000
    bf16*  featbf0 = (bf16*) (ws + 40960000);                 // N*128*2
    bf16*  featbf1 = (bf16*) (ws + 51200000);
    bf16*  featbf2 = (bf16*) (ws + 61440000);                 // f_out bf16 (kept)
    bf16*  Mbf     = (bf16*) (ws + 71680000);
    float* featf0  = (float*)(ws + 81920000);                 // N*128*4
    float* featf1  = (float*)(ws + 102400000);
    float* featf2  = (float*)(ws + 122880000);
    float* agg     = (float*)(ws + 143360000);
    bf16*  wpA     = (bf16*) (ws + 163840000);                // 128*128*2
    bf16*  wpB     = (bf16*) (ws + 163872768);
    float* bb_out  = (float*)(ws + 163905536);                // N*16*4
    float* sc_out  = (float*)(ws + 166465536);

    // ---- param layout: jax pytree sorted keys ----
    // top: bb, fe, sc ; module: l0, l1, ls ;
    // layer: W_msg, W_self, b, b_r1, b_r2, w_r1, w_r2
    auto mk = [&](size_t off, int din, int dout) {
        Lay L; size_t dd = (size_t)din * dout;
        L.Wm  = P + off;
        L.Ws  = P + off + dd;
        L.b   = P + off + 2*dd;
        L.br1 = P + off + 2*dd + dout;
        L.br2 = P + off + 2*dd + dout + 32;
        L.wr1 = P + off + 2*dd + 2*(size_t)dout + 32;
        L.wr2 = P + off + 2*dd + 2*(size_t)dout + 32 + 512;
        L.din = din; L.dout = dout;
        return L;
    };
    Lay bb_l0 = mk(0,      128, 128);
    Lay bb_l1 = mk(37664,  128, 6);
    Lay bb_ls = mk(39948,  128, 128);
    Lay fe_l0 = mk(77612,  112, 128);
    Lay fe_l1 = mk(111180, 128, 128);
    Lay fe_ls = mk(148844, 128, 128);
    Lay sc_l0 = mk(186508, 128, 128);
    Lay sc_l1 = mk(224172, 128, 16);
    Lay sc_ls = mk(229356, 128, 128);

    // ---- one GNN layer: pack -> dual WMMA GEMM -> edge scatter -> post ----
    auto run_layer = [&](const Lay& L, const bf16* inbf,
                         float* outf, bf16* outbf, const float* res,
                         int act, int dout_t) {
        const int ktiles = 4;               // K padded to 128
        const int ntiles = dout_t / 16;
        int pt = ktiles * ntiles * 32;
        k_pack_w<<<(pt + 255)/256, 256, 0, stream>>>(L.Wm, wpA, L.din, L.dout, ktiles, ntiles);
        k_pack_w<<<(pt + 255)/256, 256, 0, stream>>>(L.Ws, wpB, L.din, L.dout, ktiles, ntiles);
        int waves = (NN / 16) * ntiles;
        k_gemm_dual<<<(waves*32 + 255)/256, 256, 0, stream>>>(
            inbf, wpA, wpB, L.b, L.dout, Mbf, agg, NN, dout_t, ktiles);
        k_edge<<<2048, 256, 0, stream>>>(
            rbf, esrc, edst, Mbf, agg, L.wr1, L.br1, L.wr2, L.br2, EE, L.dout, dout_t);
        int tot = NN * dout_t;
        k_post<<<(tot + 255)/256, 256, 0, stream>>>(agg, res, outf, outbf, NN, dout_t, act);
    };

    // ---- precompute ----
    k_rbf<<<(EE + 255)/256, 256, 0, stream>>>(pos, esrc, edst, rbf, EE);
    k_build_fin<<<(NN*128 + 255)/256, 256, 0, stream>>>(f_in0, f_in1, featbf0, NN);

    // ---- fe module: DIN -> 128, 2x residual mid, 128 -> 128 (f_out) ----
    run_layer(fe_l0, featbf0, featf0, featbf1, nullptr, 1, 128);
    run_layer(fe_ls, featbf1, featf1, featbf0, featf0,  1, 128);
    run_layer(fe_ls, featbf0, featf0, featbf1, featf1,  1, 128);
    run_layer(fe_l1, featbf1, featf2, featbf2, nullptr, 0, 128);   // f_out

    // ---- bb module: f_out -> 6 (padded to 16) ----
    run_layer(bb_l0, featbf2, featf0, featbf0, nullptr, 1, 128);
    run_layer(bb_ls, featbf0, featf1, featbf1, featf0,  1, 128);
    run_layer(bb_ls, featbf1, featf0, featbf0, featf1,  1, 128);
    run_layer(bb_l1, featbf0, bb_out, nullptr, nullptr, 0, 16);

    // ---- sc module: f_out -> 16 ----
    run_layer(sc_l0, featbf2, featf0, featbf0, nullptr, 1, 128);
    run_layer(sc_ls, featbf0, featf1, featbf1, featf0,  1, 128);
    run_layer(sc_ls, featbf1, featf0, featbf0, featf1,  1, 128);
    run_layer(sc_l1, featbf0, sc_out, nullptr, nullptr, 0, 16);

    // ---- rigid-body tail ----
    k_rigid<<<(NN + 127)/128, 128, 0, stream>>>(
        bb_out, sc_out, pos, rtype, rgt, rgd, rtt, rtd, (float*)d_out, NN);
}
